// Pointnet2Encoder_89481348644932
// MI455X (gfx1250) — compile-verified, matrix operand order checked by reading, not run on hardware
//
#include <hip/hip_runtime.h>
#include <hip/hip_fp16.h>

typedef _Float16 v16h __attribute__((ext_vector_type(16)));
typedef _Float16 v8h  __attribute__((ext_vector_type(8)));
typedef float    v8f  __attribute__((ext_vector_type(8)));

// ---------------------------------------------------------------------------
// Weight conversion: w [Cin,Cout] f32 -> Wt [Cout][Kpad] f16 (transposed,
// zero-padded rows for k >= Cin), bias f32 -> f16.
// ---------------------------------------------------------------------------
__global__ void pn2_convw_kernel(const float* __restrict__ w,
                                 const float* __restrict__ b,
                                 _Float16* __restrict__ Wt,
                                 _Float16* __restrict__ bt,
                                 int Cin, int Cout, int Kpad) {
  int id = blockIdx.x * blockDim.x + threadIdx.x;
  int tot = Cout * Kpad;
  if (id < tot) {
    int n = id / Kpad, k = id % Kpad;
    Wt[id] = (k < Cin) ? (_Float16)w[(long)k * Cout + n] : (_Float16)0.0f;
  } else if (id < tot + Cout) {
    bt[id - tot] = (_Float16)b[id - tot];
  }
}

// ---------------------------------------------------------------------------
// Farthest point sampling: one workgroup per batch. Whole cloud in LDS,
// per-thread dist in registers, LDS tree-argmax (ties -> lowest index).
// ---------------------------------------------------------------------------
__global__ __launch_bounds__(1024) void pn2_fps_kernel(
    const float* __restrict__ xyz, float* __restrict__ new_xyz, int N, int S) {
  extern __shared__ __align__(16) char dynsm[];
  float* px = (float*)dynsm;
  float* py = px + N;
  float* pz = py + N;
  float* rbest = pz + N;                 // [1024]
  int*   ridx  = (int*)(rbest + 1024);   // [1024]

  const int b = blockIdx.x;
  const int tid = threadIdx.x;
  const int T = blockDim.x;
  const float* base = xyz + (long)b * N * 3;

  for (int i = tid; i < N; i += T) {
    px[i] = base[3 * i + 0];
    py[i] = base[3 * i + 1];
    pz[i] = base[3 * i + 2];
  }
  __syncthreads();

  float dist[16];
  const int P = (N + T - 1) / T;
#pragma unroll
  for (int p = 0; p < 16; ++p) dist[p] = 1e10f;

  int far = 0;
  for (int it = 0; it < S; ++it) {
    if (tid == 0) {
      float* o = new_xyz + ((long)b * S + it) * 3;
      o[0] = px[far]; o[1] = py[far]; o[2] = pz[far];
    }
    const float cx = px[far], cy = py[far], cz = pz[far];
    float best = -1.0f; int bi = 0;
    for (int p = 0; p < P; ++p) {
      int i = p * T + tid;
      if (i < N) {
        float dx = px[i] - cx, dy = py[i] - cy, dz = pz[i] - cz;
        float d = dx * dx + dy * dy + dz * dz;
        float dd = fminf(dist[p], d);
        dist[p] = dd;
        if (dd > best) { best = dd; bi = i; }
      }
    }
    rbest[tid] = best; ridx[tid] = bi;
    __syncthreads();
    for (int off = T >> 1; off > 0; off >>= 1) {
      if (tid < off) {
        float ob = rbest[tid + off]; int oi = ridx[tid + off];
        if (ob > rbest[tid] || (ob == rbest[tid] && oi < ridx[tid])) {
          rbest[tid] = ob; ridx[tid] = oi;
        }
      }
      __syncthreads();
    }
    far = ridx[0];
    __syncthreads();   // protect ridx[0] from next iteration's writes
  }
}

// ---------------------------------------------------------------------------
// Ball query, thread-per-centroid: first nsample by ascending index,
// pad with the first hit (center is always inside its own ball).
// ---------------------------------------------------------------------------
__global__ void pn2_ballq_kernel(const float* __restrict__ xyz,
                                 const float* __restrict__ new_xyz,
                                 int* __restrict__ nidx,
                                 int B, int N, int S, int ns, float r2) {
  int id = blockIdx.x * blockDim.x + threadIdx.x;
  if (id >= B * S) return;
  int b = id / S, s = id % S;
  const float* q = new_xyz + ((long)b * S + s) * 3;
  const float qx = q[0], qy = q[1], qz = q[2];
  int* out = nidx + ((long)b * S + s) * ns;
  const float* pb = xyz + (long)b * N * 3;
  int cnt = 0, first = 0;
  for (int n = 0; n < N && cnt < ns; ++n) {
    float dx = pb[3 * n + 0] - qx;
    float dy = pb[3 * n + 1] - qy;
    float dz = pb[3 * n + 2] - qz;
    if (dx * dx + dy * dy + dz * dz < r2) {
      if (cnt == 0) first = n;
      out[cnt++] = n;
    }
  }
  for (int t = cnt; t < ns; ++t) out[t] = first;
}

// ---------------------------------------------------------------------------
// One 64x16 (or 32x16) output strip: MTS row-tiles accumulated against a
// single shared B-fragment stream -> 1 global B-load pair feeds MTS WMMAs.
// ---------------------------------------------------------------------------
template <int MTS>
__device__ __forceinline__ void pn2_gemm_strip(
    const _Float16* __restrict__ X, int sx,
    _Float16* __restrict__ Y, int sy,
    const _Float16* __restrict__ Wt, const _Float16* __restrict__ bias,
    int K, int nt, int mt0, int lane) {
  const int kbase = (lane < 16) ? 0 : 8;
  const int col   = lane & 15;

  v8f acc[MTS];
  const v8f vz = {};
#pragma unroll
  for (int i = 0; i < MTS; ++i) acc[i] = vz;

  const _Float16* wrow = Wt + (long)(nt * 16 + col) * K + kbase;
  for (int k = 0; k < K; k += 32) {
    __builtin_prefetch(wrow + k + 64, 0, 1);   // -> global_prefetch_b8
    v16h bb;
    ((v8h*)&bb)[0] = *(const v8h*)(wrow + k);
    ((v8h*)&bb)[1] = *(const v8h*)(wrow + k + 16);
#pragma unroll
    for (int i = 0; i < MTS; ++i) {
      const _Float16* xrow = X + (long)((mt0 + i) * 16 + col) * sx + kbase + k;
      v16h a;
      ((v8h*)&a)[0] = *(const v8h*)(xrow);
      ((v8h*)&a)[1] = *(const v8h*)(xrow + 16);
      acc[i] = __builtin_amdgcn_wmma_f32_16x16x32_f16(
          /*neg_a=*/false, a, /*neg_b=*/false, bb,
          /*c_mod=*/(short)0, acc[i], /*reuse_a=*/false, /*reuse_b=*/false);
    }
  }

  const float bv = (float)bias[nt * 16 + col];
  const int rb = (lane < 16) ? 0 : 8;
#pragma unroll
  for (int i = 0; i < MTS; ++i) {
#pragma unroll
    for (int v = 0; v < 8; ++v) {
      float o = acc[i][v] + bv;
      o = o > 0.0f ? o : 0.0f;
      Y[(long)((mt0 + i) * 16 + rb + v) * sy + nt * 16 + col] = (_Float16)o;
    }
  }
}

// ---------------------------------------------------------------------------
// WMMA GEMM pass over a 64-row LDS tile: Y = ReLU(X @ W + b).
// X: LDS f16 [64][sx], Wt: global f16 [Nout][K] (pre-transposed, K%32==0).
// 8 waves; each wave owns full 64-row column strips (Nout>=128), or wave
// pairs split a strip 2/2 when Nout==64 so no wave idles.
// ---------------------------------------------------------------------------
__device__ __forceinline__ void pn2_gemm_relu(
    const _Float16* __restrict__ X, int sx,
    _Float16* __restrict__ Y, int sy,
    const _Float16* __restrict__ Wt, const _Float16* __restrict__ bias,
    int K, int Nout) {
  const int tid  = threadIdx.x;
  const int lane = tid & 31;
  const int wv   = tid >> 5;
  const int ct = Nout >> 4;  // column tiles
  if (ct >= 8) {
    for (int nt = wv; nt < ct; nt += 8)
      pn2_gemm_strip<4>(X, sx, Y, sy, Wt, bias, K, nt, 0, lane);
  } else {  // ct == 4: two waves per strip, 2 row-tiles each
    for (int nt = (wv >> 1); nt < ct; nt += 4)
      pn2_gemm_strip<2>(X, sx, Y, sy, Wt, bias, K, nt, (wv & 1) * 2, lane);
  }
}

// ---------------------------------------------------------------------------
// Fused: gather(rel-xyz + prev feats) -> 3x (WMMA GEMM + bias + ReLU) -> maxpool.
// One workgroup = 64 rows (a whole number of neighbor groups).
// LDS ping-pong panels: A[64][552], B[64][520] f16 (~134 KB).
// ---------------------------------------------------------------------------
#define PN2_SA 552
#define PN2_SB 520

__global__ __launch_bounds__(256) void pn2_sa_mlp_kernel(
    const float* __restrict__ xyz, const float* __restrict__ new_xyz,
    const _Float16* __restrict__ featsPrev, const int* __restrict__ nidx,
    const _Float16* __restrict__ Wt0, const _Float16* __restrict__ b0,
    const _Float16* __restrict__ Wt1, const _Float16* __restrict__ b1,
    const _Float16* __restrict__ Wt2, const _Float16* __restrict__ b2,
    _Float16* __restrict__ featsOut, float* __restrict__ outF32,
    int N, int S, int ns, int Cin, int Kpad0, int d1, int d2, int d3,
    float inv_radius) {
  extern __shared__ __align__(16) char dynsm[];
  _Float16* A  = (_Float16*)dynsm;          // [64][PN2_SA]
  _Float16* Bf = A + 64 * PN2_SA;           // [64][PN2_SB]

  const int tid = threadIdx.x;
  const long grow0 = (long)blockIdx.x * 64;
  const int Cprev = Cin - 3;

  // --- Build input panel: rel xyz (normalized) ++ gathered feats ++ zero pad
  const int total = 64 * Kpad0;
  for (int e = tid; e < total; e += 256) {
    int r = e / Kpad0, c = e % Kpad0;
    long gr = grow0 + r;
    int b = (int)(gr / ((long)S * ns));
    long rem = gr % ((long)S * ns);
    int s = (int)(rem / ns);
    int j = (int)(rem % ns);
    int n = nidx[((long)b * S + s) * ns + j];
    float v = 0.0f;
    if (c < 3) {
      v = (xyz[((long)b * N + n) * 3 + c] - new_xyz[((long)b * S + s) * 3 + c]) * inv_radius;
    } else if (c < 3 + Cprev) {
      v = (float)featsPrev[((long)b * N + n) * Cprev + (c - 3)];
    }
    A[(long)r * PN2_SA + c] = (_Float16)v;
  }
  __syncthreads();

  pn2_gemm_relu(A,  PN2_SA, Bf, PN2_SB, Wt0, b0, Kpad0, d1); __syncthreads();
  pn2_gemm_relu(Bf, PN2_SB, A,  PN2_SA, Wt1, b1, d1,    d2); __syncthreads();
  pn2_gemm_relu(A,  PN2_SA, Bf, PN2_SB, Wt2, b2, d2,    d3); __syncthreads();

  // --- Max pool over nsample (outputs are >= 0 after ReLU)
  const int ngroups = 64 / ns;
  const int totalo = ngroups * d3;
  for (int e = tid; e < totalo; e += 256) {
    int g = e / d3, ch = e % d3;
    float m = 0.0f;
    for (int j = 0; j < ns; ++j)
      m = fmaxf(m, (float)Bf[(long)(g * ns + j) * PN2_SB + ch]);
    long gcent = grow0 / ns + g;   // == b*S + s
    featsOut[gcent * d3 + ch] = (_Float16)m;
    if (outF32) outF32[gcent * d3 + ch] = m;
  }
}

// ---------------------------------------------------------------------------
// Host orchestration
// ---------------------------------------------------------------------------
extern "C" void kernel_launch(void* const* d_in, const int* in_sizes, int n_in,
                              void* d_out, int out_size, void* d_ws, size_t ws_size,
                              hipStream_t stream) {
  (void)in_sizes; (void)n_in; (void)out_size; (void)ws_size;
  const int B = 4;
  static const int   Ss[4] = {2048, 1024, 512, 256};
  static const int   NSs[4] = {64, 32, 16, 8};
  static const float Rs[4] = {0.2f, 0.4f, 0.6f, 1.2f};
  static const int CinL[4][3]  = {{3, 64, 64},   {131, 128, 128}, {259, 256, 256}, {515, 512, 512}};
  static const int CoutL[4][3] = {{64, 64, 128}, {128, 128, 256}, {256, 256, 512}, {512, 512, 512}};
  static const int KpadL[4][3] = {{32, 64, 64},  {160, 128, 128}, {288, 256, 256}, {544, 512, 512}};

  const float* pc = (const float*)d_in[0];
  float* outF = (float*)d_out;

  // bump allocator on workspace (256B aligned)
  char* ws = (char*)d_ws;
  size_t off = 0;
  auto alloc = [&](size_t bytes) -> void* {
    off = (off + 255) & ~(size_t)255;
    void* p = ws + off;
    off += bytes;
    return p;
  };

  float* newxyz[4];
  for (int s = 0; s < 4; ++s) newxyz[s] = (float*)alloc((size_t)B * Ss[s] * 3 * sizeof(float));
  int* nidxbuf = (int*)alloc((size_t)B * 2048 * 64 * sizeof(int));            // max across stages
  _Float16* featsA = (_Float16*)alloc((size_t)1048576 * sizeof(_Float16));    // max feat panel
  _Float16* featsB = (_Float16*)alloc((size_t)1048576 * sizeof(_Float16));
  _Float16 *WtP[4][3], *btP[4][3];
  for (int s = 0; s < 4; ++s)
    for (int l = 0; l < 3; ++l) {
      WtP[s][l] = (_Float16*)alloc((size_t)CoutL[s][l] * KpadL[s][l] * sizeof(_Float16));
      btP[s][l] = (_Float16*)alloc((size_t)CoutL[s][l] * sizeof(_Float16));
    }

  // 1) convert + transpose + pad all weights to f16
  for (int s = 0; s < 4; ++s)
    for (int l = 0; l < 3; ++l) {
      const float* w  = (const float*)d_in[1 + s * 6 + l * 2];
      const float* bb = (const float*)d_in[2 + s * 6 + l * 2];
      int tot = CoutL[s][l] * KpadL[s][l] + CoutL[s][l];
      pn2_convw_kernel<<<(tot + 255) / 256, 256, 0, stream>>>(
          w, bb, WtP[s][l], btP[s][l], CinL[s][l], CoutL[s][l], KpadL[s][l]);
    }

  // 2) four SA stages
  const float* cur_xyz = pc;
  const _Float16* cur_feats = nullptr;
  int N = 16384;
  const size_t mlp_lds = (size_t)(64 * PN2_SA + 64 * PN2_SB) * sizeof(_Float16);

  for (int s = 0; s < 4; ++s) {
    const int S = Ss[s], ns = NSs[s];
    const float r = Rs[s];

    size_t fps_lds = (size_t)(3 * N + 2 * 1024) * sizeof(float);
    pn2_fps_kernel<<<B, 1024, fps_lds, stream>>>(cur_xyz, newxyz[s], N, S);

    pn2_ballq_kernel<<<(B * S + 255) / 256, 256, 0, stream>>>(
        cur_xyz, newxyz[s], nidxbuf, B, N, S, ns, r * r);

    _Float16* fOut = (s & 1) ? featsB : featsA;
    float* outFeats = (s == 3) ? (outF + (size_t)B * 256 * 3) : nullptr;
    int blocks = (B * S * ns) / 64;
    pn2_sa_mlp_kernel<<<blocks, 256, mlp_lds, stream>>>(
        cur_xyz, newxyz[s], cur_feats, nidxbuf,
        WtP[s][0], btP[s][0], WtP[s][1], btP[s][1], WtP[s][2], btP[s][2],
        fOut, outFeats,
        N, S, ns, CinL[s][0], KpadL[s][0],
        CoutL[s][0], CoutL[s][1], CoutL[s][2], 1.0f / r);

    cur_xyz = newxyz[s];
    cur_feats = fOut;
    N = S;
  }

  // 3) final xyz -> d_out[0 : B*256*3]
  hipMemcpyAsync(outF, newxyz[3], (size_t)B * 256 * 3 * sizeof(float),
                 hipMemcpyDeviceToDevice, stream);
}
